// get_model_50431505989771
// MI455X (gfx1250) — matrix operand order repaired and checked by reading, not compiled, not run on hardware
//
#include <hip/hip_runtime.h>
#include <math.h>

typedef _Float16 H;
typedef __attribute__((ext_vector_type(16))) _Float16 v16h;
typedef __attribute__((ext_vector_type(8)))  _Float16 v8h;
typedef __attribute__((ext_vector_type(8)))  float    v8f;

#define CMAX 512            // max conv output channels
#define NCLS 13

__device__ inline v16h cat8(v8h a, v8h b){
  v16h r;
#pragma unroll
  for (int i = 0; i < 8; ++i){ r[i] = a[i]; r[8+i] = b[i]; }
  return r;
}

// X activation panel layout: element (c, m) at ((c>>5)*M + m)*32 + (c&31)
__device__ inline size_t xidx(int c, size_t m, size_t M){
  return ((size_t)(c >> 5) * M + m) * 32 + (c & 31);
}

// ---------------------------------------------------------------------------
// Weight staging: f32 (cout x cin) -> f16 zero-padded (cout_pad x cin_pad)
// ---------------------------------------------------------------------------
__global__ void stage_weights_kernel(const float* __restrict__ W, H* __restrict__ Wp,
                                     int cout, int cin, int cout_pad, int cin_pad){
  int i = blockIdx.x * 256 + threadIdx.x;
  int total = cout_pad * cin_pad;
  if (i >= total) return;
  int co = i / cin_pad, k = i % cin_pad;
  Wp[i] = (co < cout && k < cin) ? (H)W[(size_t)co * cin + k] : (H)0.f;
}

// ---------------------------------------------------------------------------
// Farthest point sampling: one block (256 thr = 8 waves) per batch.
// Writes selected coordinates as (B, npoint, 3).
// ---------------------------------------------------------------------------
__global__ void fps_kernel(const float* __restrict__ xyz, long long bs, int cs, int ps,
                           int n, int npoint, float* __restrict__ out){
  __shared__ float dist[4096];
  __shared__ float rv[8];
  __shared__ int   ri[8];
  __shared__ float cc[3];
  __shared__ int   cur;
  const int b = blockIdx.x, tid = threadIdx.x;
  const float* p = xyz + (size_t)b * bs;
  for (int i = tid; i < n; i += 256) dist[i] = 1e10f;
  if (tid == 0) cur = 0;
  __syncthreads();
  for (int t = 0; t < npoint; ++t){
    if (tid == 0){
      int c0 = cur;
      cc[0] = p[0*cs + c0*ps];
      cc[1] = p[1*cs + c0*ps];
      cc[2] = p[2*cs + c0*ps];
      float* o = out + ((size_t)b * npoint + t) * 3;
      o[0] = cc[0]; o[1] = cc[1]; o[2] = cc[2];
    }
    __syncthreads();
    float cx = cc[0], cy = cc[1], cz = cc[2];
    float bv = -1.0f; int bi = 0;
    for (int i = tid; i < n; i += 256){
      float dx = p[0*cs + i*ps] - cx;
      float dy = p[1*cs + i*ps] - cy;
      float dz = p[2*cs + i*ps] - cz;
      float d  = dx*dx + dy*dy + dz*dz;
      d = fminf(dist[i], d);
      dist[i] = d;
      if (d > bv){ bv = d; bi = i; }
    }
    for (int off = 16; off > 0; off >>= 1){
      float ov = __shfl_down(bv, off, 32);
      int   oi = __shfl_down(bi, off, 32);
      if (ov > bv || (ov == bv && oi < bi)){ bv = ov; bi = oi; }
    }
    if ((tid & 31) == 0){ rv[tid >> 5] = bv; ri[tid >> 5] = bi; }
    __syncthreads();
    if (tid == 0){
      float v = rv[0]; int ix = ri[0];
      for (int j = 1; j < 8; ++j)
        if (rv[j] > v || (rv[j] == v && ri[j] < ix)){ v = rv[j]; ix = ri[j]; }
      cur = ix;
    }
    __syncthreads();
  }
}

// ---------------------------------------------------------------------------
// Ball query: first k ascending indices with sqdist <= r^2; pad with first.
// ---------------------------------------------------------------------------
__global__ void ball_query_kernel(const float* __restrict__ xyz, long long bs, int cs, int ps,
                                  int n, const float* __restrict__ new_xyz, int np,
                                  float r2, int k, int* __restrict__ gidx, int total){
  int t = blockIdx.x * 256 + threadIdx.x;
  if (t >= total) return;
  int b = t / np, q = t % np;
  const float* nx = new_xyz + ((size_t)b * np + q) * 3;
  float qx = nx[0], qy = nx[1], qz = nx[2];
  const float* p = xyz + (size_t)b * bs;
  int* out = gidx + (size_t)t * k;
  int cnt = 0, first = -1;
  for (int i = 0; i < n && cnt < k; ++i){
    float dx = p[0*cs + i*ps] - qx;
    float dy = p[1*cs + i*ps] - qy;
    float dz = p[2*cs + i*ps] - qz;
    float d2 = dx*dx + dy*dy + dz*dz;
    if (d2 <= r2){
      if (first < 0) first = i;
      out[cnt++] = i;
    }
  }
  if (first < 0) first = 0;
  for (int j = cnt; j < k; ++j) out[j] = first;
}

// ---------------------------------------------------------------------------
// SA gather -> f16 X panels (cin_pad x M), M = B*np*k, m=((b*np+q)*k+s).
// Rows: cpts feature channels, 3 rel-xyz channels, zeros to cin_pad.
// ---------------------------------------------------------------------------
__global__ void sa_gather_kernel(const float* __restrict__ feat, long long f_bs, int f_cs,
                                 const float* __restrict__ sxyz, long long x_bs, int x_cs, int x_ps,
                                 const float* __restrict__ new_xyz, const int* __restrict__ gidx,
                                 int np, int k, int cpts, int cin_pad,
                                 H* __restrict__ X, size_t M){
  size_t m = (size_t)blockIdx.x * 256 + threadIdx.x;
  if (m >= M) return;
  size_t t = m / k;
  int q = (int)(t % np);
  int b = (int)(t / np);
  int i = gidx[m];
  const float* nx = new_xyz + ((size_t)b * np + q) * 3;
  const float* p  = sxyz + (size_t)b * x_bs;
  float rx = p[0*x_cs + i*x_ps] - nx[0];
  float ry = p[1*x_cs + i*x_ps] - nx[1];
  float rz = p[2*x_cs + i*x_ps] - nx[2];
  const float* fb = feat + (size_t)b * f_bs;
  for (int c = 0; c < cpts; ++c)
    X[xidx(c, m, M)] = (H)fb[(size_t)c * f_cs + i];
  X[xidx(cpts + 0, m, M)] = (H)rx;
  X[xidx(cpts + 1, m, M)] = (H)ry;
  X[xidx(cpts + 2, m, M)] = (H)rz;
  for (int c = cpts + 3; c < cin_pad; ++c)
    X[xidx(c, m, M)] = (H)0.f;
}

// ---------------------------------------------------------------------------
// GEMM: Y(cout x M) = Wp(cout_pad x cin_pad) * Xp + bias via
// v_wmma_f32_16x16x32_f16.  Per-wave 16x16 tile; 8 waves = 16co x 128m.
// Software-pipelined: panel k+1 fragment loads issued before wmma(k), so the
// matrix op only waits for the previous iteration's loads (s_wait_loadcnt>0).
// Also accumulates per-channel sum/sumsq for BN.
// ---------------------------------------------------------------------------
__global__ void gemm_bn_stats_kernel(const H* __restrict__ Wp, const H* __restrict__ Xp,
                                     const float* __restrict__ bias,
                                     float* __restrict__ Y, float* __restrict__ stats,
                                     int cout, int cin_pad, size_t M){
  int lane = threadIdx.x & 31;
  int wave = threadIdx.x >> 5;
  size_t m0 = (size_t)blockIdx.x * 128 + (size_t)wave * 16;
  int co0 = blockIdx.y * 16;

  int row = co0 + (lane & 15);
  int kb  = (lane >> 4) * 16;       // A: 0 or 16
  int kb2 = (lane >> 4) * 16;       // B: 0 or 16
  size_t col = m0 + (lane & 15);

  const v8h* pa = (const v8h*)(Wp + (size_t)row * cin_pad + kb);
  const v8h* pb = (const v8h*)(Xp + col * 32 + kb2);
  const size_t bstep = (size_t)M * 4;   // one panel in v8h units

  v8f c = {};
  int npanel = cin_pad >> 5;

  // prologue: fragments for panel 0
  v8h a0 = pa[0], a1 = pa[1];           // K = kb..kb+15 (contiguous)
  v8h b0 = pb[0], b1 = pb[1];           // K = kb2..kb2+15 (contiguous)

  for (int kp = 0; kp < npanel - 1; ++kp){
    pa += 4;                            // +32 halves
    pb += bstep;
    __builtin_prefetch(pb + bstep, 0, 1);
    v8h na0 = pa[0], na1 = pa[1];       // next-panel loads in flight...
    v8h nb0 = pb[0], nb1 = pb[1];
    c = __builtin_amdgcn_wmma_f32_16x16x32_f16(false, cat8(a0, a1), false, cat8(b0, b1),
                                               (short)0, c, false, false);
    a0 = na0; a1 = na1; b0 = nb0; b1 = nb1;
  }
  c = __builtin_amdgcn_wmma_f32_16x16x32_f16(false, cat8(a0, a1), false, cat8(b0, b1),
                                             (short)0, c, false, false);

  size_t m = m0 + (lane & 15);
  int hf = lane >> 4;
#pragma unroll
  for (int v = 0; v < 8; ++v){
    int co = co0 + v + hf * 8;
    bool ok = co < cout;
    float val = ok ? (c[v] + bias[co]) : 0.f;
    if (ok) Y[(size_t)co * M + m] = val;
    float s = val, s2 = val * val;
#pragma unroll
    for (int off = 1; off < 16; off <<= 1){
      s  += __shfl_xor(s,  off, 32);
      s2 += __shfl_xor(s2, off, 32);
    }
    if (((lane & 15) == 0) && ok){
      atomicAdd(&stats[co],        s);
      atomicAdd(&stats[CMAX + co], s2);
    }
  }
}

// BN finalize: per-channel scale/shift
__global__ void bn_finalize_kernel(float* __restrict__ stats, const float* __restrict__ gamma,
                                   const float* __restrict__ beta, int cout, size_t M){
  int c = blockIdx.x * 64 + threadIdx.x;
  if (c >= cout) return;
  float inv  = 1.f / (float)M;
  float mean = stats[c] * inv;
  float var  = stats[CMAX + c] * inv - mean * mean;
  float sc   = gamma[c] * rsqrtf(var + 1e-5f);
  stats[2*CMAX + c] = sc;
  stats[3*CMAX + c] = beta[c] - mean * sc;
}

// BN apply + ReLU -> f16 X panels (gridDim.y = cout_pad32; pad rows zeroed)
__global__ void bn_apply_f16_kernel(const float* __restrict__ Y, const float* __restrict__ stats,
                                    H* __restrict__ Xo, size_t M, int cout){
  int c = blockIdx.y;
  size_t m = (size_t)blockIdx.x * 256 + threadIdx.x;
  float v = 0.f;
  if (c < cout)
    v = fmaxf(Y[(size_t)c * M + m] * stats[2*CMAX + c] + stats[3*CMAX + c], 0.f);
  Xo[xidx(c, m, M)] = (H)v;
}

// BN apply + ReLU -> f32 points layout (B, C, n), m = b*n + i
__global__ void bn_apply_points_kernel(const float* __restrict__ Y, const float* __restrict__ stats,
                                       float* __restrict__ P, size_t M, int n, int C){
  int c = blockIdx.y;
  size_t m = (size_t)blockIdx.x * 256 + threadIdx.x;
  float v = fmaxf(Y[(size_t)c * M + m] * stats[2*CMAX + c] + stats[3*CMAX + c], 0.f);
  int b = (int)(m / n), i = (int)(m % n);
  P[((size_t)b * C + c) * n + i] = v;
}

// BN apply + ReLU + max over k neighbors -> points (B, Ctot, np) at coff
__global__ void bn_maxpool_kernel(const float* __restrict__ Y, const float* __restrict__ stats,
                                  float* __restrict__ P, int B_, int np, int k,
                                  int C, int Ctot, int coff, size_t M){
  int idx = blockIdx.x * 256 + threadIdx.x;
  int total = B_ * C * np;
  if (idx >= total) return;
  int b = idx / (C * np);
  int r = idx % (C * np);
  int c = r / np, q = r % np;
  float sc = stats[2*CMAX + c], sh = stats[3*CMAX + c];
  size_t base = ((size_t)b * np + q) * (size_t)k;
  float mx = -3.4e38f;
  for (int s = 0; s < k; ++s)
    mx = fmaxf(mx, Y[(size_t)c * M + base + s] * sc + sh);
  mx = fmaxf(mx, 0.f);
  P[((size_t)b * Ctot + coff + c) * np + q] = mx;
}

// ---------------------------------------------------------------------------
// Feature propagation: 3-NN inverse-distance interp + concat -> f16 X panels.
// (C1 + C2 is a multiple of 32 for every FP level.)
// ---------------------------------------------------------------------------
__global__ void fp_interp_kernel(const float* __restrict__ xyz1, long long bs1, int cs1, int ps1, int n1,
                                 const float* __restrict__ xyz2, int n2,
                                 const float* __restrict__ P1, int C1,
                                 const float* __restrict__ P2, int C2,
                                 H* __restrict__ X, size_t M){
  size_t m = (size_t)blockIdx.x * 256 + threadIdx.x;
  if (m >= M) return;
  int b = (int)(m / n1), i = (int)(m % n1);
  const float* x1 = xyz1 + (size_t)b * bs1;
  float px = x1[0*cs1 + i*ps1], py = x1[1*cs1 + i*ps1], pz = x1[2*cs1 + i*ps1];
  float w0 = 1.f, w1 = 0.f, w2 = 0.f;
  int i0 = 0, i1 = 0, i2 = 0;
  if (n2 > 1){
    float d0 = 1e30f, d1 = 1e30f, d2 = 1e30f;
    const float* x2 = xyz2 + (size_t)b * n2 * 3;
    for (int j = 0; j < n2; ++j){
      float dx = x2[j*3+0] - px, dy = x2[j*3+1] - py, dz = x2[j*3+2] - pz;
      float d = dx*dx + dy*dy + dz*dz;
      if (d < d0){ d2=d1; i2=i1; d1=d0; i1=i0; d0=d; i0=j; }
      else if (d < d1){ d2=d1; i2=i1; d1=d; i1=j; }
      else if (d < d2){ d2=d; i2=j; }
    }
    w0 = 1.f/(d0 + 1e-8f); w1 = 1.f/(d1 + 1e-8f); w2 = 1.f/(d2 + 1e-8f);
    float ws = w0 + w1 + w2;
    w0 /= ws; w1 /= ws; w2 /= ws;
  }
  if (P1){
    const float* p1b = P1 + (size_t)b * C1 * n1;
    for (int c = 0; c < C1; ++c)
      X[xidx(c, m, M)] = (H)p1b[(size_t)c * n1 + i];
  }
  const float* p2b = P2 + (size_t)b * C2 * n2;
  for (int c = 0; c < C2; ++c){
    const float* row = p2b + (size_t)c * n2;
    X[xidx(C1 + c, m, M)] = (H)(w0 * row[i0] + w1 * row[i1] + w2 * row[i2]);
  }
}

// ---------------------------------------------------------------------------
// Head: logits = W2(13x128) * x + b2, then log_softmax.  X in panel layout.
// ---------------------------------------------------------------------------
__global__ void head_kernel(const H* __restrict__ X, const float* __restrict__ W2,
                            const float* __restrict__ b2, float* __restrict__ out, size_t M){
  __shared__ float w2s[NCLS * 128];
  for (int t = threadIdx.x; t < NCLS * 128; t += 256) w2s[t] = W2[t];
  __syncthreads();
  size_t m = (size_t)blockIdx.x * 256 + threadIdx.x;
  if (m >= M) return;
  float acc[NCLS];
#pragma unroll
  for (int o = 0; o < NCLS; ++o) acc[o] = b2[o];
  for (int p = 0; p < 4; ++p){                      // 4 panels of 32 channels
    const H* base = X + ((size_t)p * M + m) * 32;
    for (int j = 0; j < 32; ++j){
      float xv = (float)base[j];
      int c = p * 32 + j;
#pragma unroll
      for (int o = 0; o < NCLS; ++o) acc[o] += w2s[o*128 + c] * xv;
    }
  }
  float mx = -3.4e38f;
#pragma unroll
  for (int o = 0; o < NCLS; ++o) mx = fmaxf(mx, acc[o]);
  float ssum = 0.f;
#pragma unroll
  for (int o = 0; o < NCLS; ++o) ssum += expf(acc[o] - mx);
  float ls = logf(ssum);
#pragma unroll
  for (int o = 0; o < NCLS; ++o) out[m * NCLS + o] = acc[o] - mx - ls;
}

// ===========================================================================
// Host orchestration
// ===========================================================================
extern "C" void kernel_launch(void* const* d_in, const int* in_sizes, int n_in,
                              void* d_out, int out_size, void* d_ws, size_t ws_size,
                              hipStream_t stream){
  (void)in_sizes; (void)n_in; (void)out_size; (void)ws_size;
  const int B = 8, N = 4096;
  const float* xyz_in = (const float*)d_in[0];

  // ---- workspace arena ----
  char* ws = (char*)d_ws;
  size_t off = 0;
  auto arena = [&](size_t bytes)->char*{
    char* p = ws + off;
    off += (bytes + 255) & ~(size_t)255;
    return p;
  };
  float* Y    = (float*)arena(68ull << 20);               // f32 GEMM output
  H*     X0   = (H*)    arena(36ull << 20);               // ping (panel layout)
  H*     X1   = (H*)    arena(36ull << 20);               // pong (panel layout)
  H*     W16  = (H*)    arena(2ull  << 20);               // padded f16 weights
  float* stats= (float*)arena(4 * CMAX * sizeof(float));  // sum|sumsq|scale|shift
  float* nx1  = (float*)arena((size_t)B*1024*3*4);
  float* nx2  = (float*)arena((size_t)B*256*3*4);
  float* nx3  = (float*)arena((size_t)B*64*3*4);
  float* nx4  = (float*)arena((size_t)B*16*3*4);
  int*   gidx = (int*)  arena((size_t)B*1024*32*4);
  float* pts1 = (float*)arena((size_t)B*96*1024*4);
  float* pts2 = (float*)arena((size_t)B*256*256*4);
  float* pts3 = (float*)arena((size_t)B*512*64*4);
  float* pts4 = (float*)arena((size_t)B*1024*16*4);
  float* fpo3 = (float*)arena((size_t)B*256*64*4);
  float* fpo2 = (float*)arena((size_t)B*256*256*4);
  float* fpo1 = (float*)arena((size_t)B*128*1024*4);

  auto pad16 = [](int x){ return (x + 15) & ~15; };
  auto pad32 = [](int x){ return (x + 31) & ~31; };

  // ---- parameter tables (insertion order of setup_inputs) ----
  struct Layer { const float *W,*b,*g,*be; int cout,cin; };
  int pi = 1;
  auto grab = [&](int cout, int cin)->Layer{
    Layer L{(const float*)d_in[pi], (const float*)d_in[pi+1],
            (const float*)d_in[pi+2], (const float*)d_in[pi+3], cout, cin};
    pi += 4; return L;
  };
  static const int sa_np[4]   = {1024, 256, 64, 16};
  static const int sa_nsrc[4] = {4096, 1024, 256, 64};
  static const float sa_rad[4][2] = {{0.05f,0.1f},{0.1f,0.2f},{0.2f,0.4f},{0.4f,0.8f}};
  static const int sa_k[2] = {16, 32};
  static const int sa_cinv[4] = {12, 99, 259, 515};
  static const int sa_dims[4][2][3] = {
    {{16,16,32},{32,32,64}},
    {{64,64,128},{64,96,128}},
    {{128,196,256},{128,196,256}},
    {{256,256,512},{256,384,512}}};
  Layer sa[4][2][3];
  for (int l = 0; l < 4; l++)
    for (int br = 0; br < 2; br++){
      int cin = sa_cinv[l];
      for (int li = 0; li < 3; li++){ sa[l][br][li] = grab(sa_dims[l][br][li], cin); cin = sa_dims[l][br][li]; }
    }
  static const int fp_cin[4]    = {1536, 512, 352, 128};
  static const int fp_nl[4]     = {2, 2, 2, 3};
  static const int fp_dims[4][3]= {{256,256,0},{256,256,0},{256,128,0},{128,128,128}};
  Layer fp[4][3];
  for (int f = 0; f < 4; f++){
    int cin = fp_cin[f];
    for (int li = 0; li < fp_nl[f]; li++){ fp[f][li] = grab(fp_dims[f][li], cin); cin = fp_dims[f][li]; }
  }
  Layer hc1 = grab(128, 128);
  const float* W2 = (const float*)d_in[pi];
  const float* b2 = (const float*)d_in[pi+1];

  // ---- conv (stage weights + GEMM + BN stats) ----
  auto conv = [&](const Layer& L, const H* Xin, size_t M){
    int cop = pad16(L.cout), cip = pad32(L.cin);
    int wn = cop * cip;
    stage_weights_kernel<<<(wn + 255)/256, 256, 0, stream>>>(L.W, W16, L.cout, L.cin, cop, cip);
    hipMemsetAsync(stats, 0, 2 * CMAX * sizeof(float), stream);
    dim3 g((unsigned)(M / 128), (unsigned)(cop / 16));
    gemm_bn_stats_kernel<<<g, 256, 0, stream>>>(W16, Xin, L.b, Y, stats, L.cout, cip, M);
    bn_finalize_kernel<<<(L.cout + 63)/64, 64, 0, stream>>>(stats, L.g, L.be, L.cout, M);
  };

  // ---- set-abstraction levels ----
  const float* feat[4]  = {xyz_in, pts1, pts2, pts3};
  long long feat_bs[4]  = {(long long)9*N, (long long)96*1024, (long long)256*256, (long long)512*64};
  int feat_cs[4]        = {N, 1024, 256, 64};
  int sa_cpts[4]        = {9, 96, 256, 512};
  const float* sxyz[4]  = {xyz_in, nx1, nx2, nx3};
  long long x_bs[4]     = {(long long)9*N, 1024*3, 256*3, 64*3};
  int x_cs[4]           = {N, 1, 1, 1};
  int x_ps[4]           = {1, 3, 3, 3};
  float* nxo[4]         = {nx1, nx2, nx3, nx4};
  float* ptso[4]        = {pts1, pts2, pts3, pts4};
  int Ctot[4]           = {96, 256, 512, 1024};

  for (int l = 0; l < 4; l++){
    int np = sa_np[l], nsrc = sa_nsrc[l];
    fps_kernel<<<B, 256, 0, stream>>>(sxyz[l], x_bs[l], x_cs[l], x_ps[l], nsrc, np, nxo[l]);
    int coff = 0;
    for (int br = 0; br < 2; br++){
      int k = sa_k[br];
      float r = sa_rad[l][br];
      int tq = B * np;
      ball_query_kernel<<<(tq + 255)/256, 256, 0, stream>>>(sxyz[l], x_bs[l], x_cs[l], x_ps[l],
                                                            nsrc, nxo[l], np, r*r, k, gidx, tq);
      size_t M = (size_t)B * np * k;
      sa_gather_kernel<<<(unsigned)(M / 256), 256, 0, stream>>>(feat[l], feat_bs[l], feat_cs[l],
                                                                sxyz[l], x_bs[l], x_cs[l], x_ps[l],
                                                                nxo[l], gidx, np, k, sa_cpts[l],
                                                                pad32(sa_cinv[l]), X0, M);
      const H* xin = X0; H* xout = X1;
      for (int li = 0; li < 3; li++){
        const Layer& L = sa[l][br][li];
        conv(L, xin, M);
        if (li < 2){
          bn_apply_f16_kernel<<<dim3((unsigned)(M/256), pad32(L.cout)), 256, 0, stream>>>(Y, stats, xout, M, L.cout);
          const H* t = xin; xin = xout; xout = (H*)t;
        } else {
          int total = B * L.cout * np;
          bn_maxpool_kernel<<<(total + 255)/256, 256, 0, stream>>>(Y, stats, ptso[l], B, np, k,
                                                                   L.cout, Ctot[l], coff, M);
        }
      }
      coff += sa[l][br][2].cout;
    }
  }

  // ---- feature propagation ----
  { // fp4: l3(64) <- l4(16)
    size_t M = (size_t)B * 64;
    fp_interp_kernel<<<(unsigned)(M/256), 256, 0, stream>>>(nx3, 64*3, 1, 3, 64,
        nx4, 16, pts3, 512, pts4, 1024, X0, M);
    conv(fp[0][0], X0, M);
    bn_apply_f16_kernel<<<dim3((unsigned)(M/256), 256), 256, 0, stream>>>(Y, stats, X1, M, 256);
    conv(fp[0][1], X1, M);
    bn_apply_points_kernel<<<dim3((unsigned)(M/256), 256), 256, 0, stream>>>(Y, stats, fpo3, M, 64, 256);
  }
  { // fp3: l2(256) <- l3'(64)
    size_t M = (size_t)B * 256;
    fp_interp_kernel<<<(unsigned)(M/256), 256, 0, stream>>>(nx2, 256*3, 1, 3, 256,
        nx3, 64, pts2, 256, fpo3, 256, X0, M);
    conv(fp[1][0], X0, M);
    bn_apply_f16_kernel<<<dim3((unsigned)(M/256), 256), 256, 0, stream>>>(Y, stats, X1, M, 256);
    conv(fp[1][1], X1, M);
    bn_apply_points_kernel<<<dim3((unsigned)(M/256), 256), 256, 0, stream>>>(Y, stats, fpo2, M, 256, 256);
  }
  { // fp2: l1(1024) <- l2'(256)
    size_t M = (size_t)B * 1024;
    fp_interp_kernel<<<(unsigned)(M/256), 256, 0, stream>>>(nx1, 1024*3, 1, 3, 1024,
        nx2, 256, pts1, 96, fpo2, 256, X0, M);
    conv(fp[2][0], X0, M);
    bn_apply_f16_kernel<<<dim3((unsigned)(M/256), 256), 256, 0, stream>>>(Y, stats, X1, M, 256);
    conv(fp[2][1], X1, M);
    bn_apply_points_kernel<<<dim3((unsigned)(M/256), 128), 256, 0, stream>>>(Y, stats, fpo1, M, 1024, 128);
  }
  { // fp1: l0(4096) <- l1'(1024), then head
    size_t M = (size_t)B * N;
    fp_interp_kernel<<<(unsigned)(M/256), 256, 0, stream>>>(xyz_in, (long long)9*N, N, 1, N,
        nx1, 1024, nullptr, 0, fpo1, 128, X0, M);
    conv(fp[3][0], X0, M);
    bn_apply_f16_kernel<<<dim3((unsigned)(M/256), 128), 256, 0, stream>>>(Y, stats, X1, M, 128);
    conv(fp[3][1], X1, M);
    bn_apply_f16_kernel<<<dim3((unsigned)(M/256), 128), 256, 0, stream>>>(Y, stats, X0, M, 128);
    conv(fp[3][2], X0, M);
    bn_apply_f16_kernel<<<dim3((unsigned)(M/256), 128), 256, 0, stream>>>(Y, stats, X1, M, 128);
    conv(hc1, X1, M);
    bn_apply_f16_kernel<<<dim3((unsigned)(M/256), 128), 256, 0, stream>>>(Y, stats, X0, M, 128);
    head_kernel<<<(unsigned)(M/256), 256, 0, stream>>>(X0, W2, b2, (float*)d_out, M);
  }

  // ---- second output: l4_points (B,1024,16) ----
  hipMemcpyAsync((float*)d_out + (size_t)B * N * NCLS, pts4,
                 (size_t)B * 1024 * 16 * sizeof(float),
                 hipMemcpyDeviceToDevice, stream);
}
